// SparseCINCochainConv_20203526161168
// MI455X (gfx1250) — matrix operand order repaired and checked by reading, not compiled, not run on hardware
//
#include <hip/hip_runtime.h>
#include <hip/hip_bf16.h>

typedef __attribute__((ext_vector_type(16))) __bf16 v16bf;
typedef __attribute__((ext_vector_type(8)))  __bf16 bf16x8;
typedef __attribute__((ext_vector_type(8)))  float  v8f;

#define HD 64

// ---------------------------------------------------------------------------
// agg_up = (1+eps1)*x ; agg_b = (1+eps2)*x
// ---------------------------------------------------------------------------
__global__ void init_agg_kernel(const float* __restrict__ x,
                                const float* __restrict__ eps1,
                                const float* __restrict__ eps2,
                                float* __restrict__ aggU,
                                float* __restrict__ aggB,
                                int total) {
    int i = blockIdx.x * blockDim.x + threadIdx.x;
    if (i < total) {
        float e1 = 1.0f + eps1[0];
        float e2 = 1.0f + eps2[0];
        float v = x[i];
        aggU[i] = e1 * v;
        aggB[i] = e2 * v;
    }
}

__global__ void zero_kernel(float* __restrict__ p, int n) {
    int i = blockIdx.x * blockDim.x + threadIdx.x;
    if (i < n) p[i] = 0.0f;
}

// ---------------------------------------------------------------------------
// agg[dst[e]] += feat[src[e]]   (16 threads per edge, float4 each)
// ---------------------------------------------------------------------------
__global__ void scatter_kernel(const float* __restrict__ feat,
                               const int* __restrict__ src,
                               const int* __restrict__ dst,
                               int E, float* __restrict__ agg) {
    long long g = (long long)blockIdx.x * blockDim.x + threadIdx.x;
    long long e = g >> 4;
    int sub = (int)(g & 15);
    if (e < (long long)E) {
        int s = src[e];
        int d = dst[e];
        float4 v = *(const float4*)(feat + (long long)s * HD + sub * 4);
        float* o = agg + (long long)d * HD + sub * 4;
        atomicAdd(o + 0, v.x);
        atomicAdd(o + 1, v.y);
        atomicAdd(o + 2, v.z);
        atomicAdd(o + 3, v.w);
    }
}

// ---------------------------------------------------------------------------
// Fragment builders (contiguous k order: element s of lane half h holds k = 16h + s, wave32).
// ---------------------------------------------------------------------------
__device__ __forceinline__ v16bf load_frag_a(const __bf16* aRow, int ks, int laneH) {
    // lanes 0-15: rows M=0..15, K=ks..ks+15
    // lanes 16-31: same rows,   K=ks+16..ks+31
    bf16x8 lo = *(const bf16x8*)(aRow + ks + laneH * 16);
    bf16x8 hi = *(const bf16x8*)(aRow + ks + laneH * 16 + 8);
    v16bf v;
    #pragma unroll
    for (int j = 0; j < 8; ++j) { v[j] = lo[j]; v[8 + j] = hi[j]; }
    return v;
}

__device__ __forceinline__ v16bf load_frag_b(const __bf16* bp) {
    // lane = column n; lanes 0-15 K=ks..ks+15, lanes 16-31 K=ks+16..ks+31
    bf16x8 b0 = *(const bf16x8*)(bp);
    bf16x8 b1 = *(const bf16x8*)(bp + 8);
    v16bf v;
    #pragma unroll
    for (int j = 0; j < 8; ++j) { v[j] = b0[j]; v[8 + j] = b1[j]; }
    return v;
}

// ---------------------------------------------------------------------------
// Fused (optional BN+ReLU of input) -> bf16 -> WMMA GEMM -> +bias -> f32 out.
// out[n][64] = act(in)[n][K] @ W[K][64] + bias
// act(v, k) = coef ? relu(coef[k]*v + coef[64+k]) : v
// K is a compile-time 64 or 128; K=128 reads its second half from in1/coef1.
// Block: 256 threads = 8 waves; each wave computes a 16-row x 64-col strip
// as 4 accumulators, K fully unrolled so B fragments hoist into registers.
// ---------------------------------------------------------------------------
template <int K>
__global__ __launch_bounds__(256)
void gemm_bn_wmma_kernel(const float* __restrict__ in0,
                         const float* __restrict__ coef0,
                         const float* __restrict__ in1,
                         const float* __restrict__ coef1,
                         const float* __restrict__ W,
                         const float* __restrict__ bias,
                         float* __restrict__ out,
                         int n_rows) {
    __shared__ __align__(16) __bf16 sW[64 * K];   // sW[n*K + k] = W[k][n]
    __shared__ __align__(16) __bf16 sA[128 * K];  // sA[r*K + k]
    const int t = threadIdx.x;

    // Stage transposed weights (coalesced global read).
    #pragma unroll 4
    for (int idx = t; idx < K * 64; idx += 256) {
        int k = idx >> 6;
        int n = idx & 63;
        sW[n * K + k] = (__bf16)W[idx];
    }

    // Stage activation tile, applying previous layer's BN+ReLU on the fly.
    const int R = blockIdx.x * 128;
    #pragma unroll 4
    for (int idx = t; idx < 128 * K; idx += 256) {
        int r = idx / K;          // K is constexpr power of two -> shift
        int k = idx & (K - 1);
        int grow = R + r;
        float v = 0.0f;
        if (grow < n_rows) {
            if (k < 64) {
                v = in0[(long long)grow * 64 + k];
                if (coef0) v = fmaxf(coef0[k] * v + coef0[64 + k], 0.0f);
            } else {
                int k2 = k - 64;
                v = in1[(long long)grow * 64 + k2];
                if (coef1) v = fmaxf(coef1[k2] * v + coef1[64 + k2], 0.0f);
            }
        }
        sA[r * K + k] = (__bf16)v;
    }
    __syncthreads();

    const int wave  = t >> 5;
    const int lane  = t & 31;
    const int laneM = lane & 15;   // row (A/D) or col (B/D) within 16x16 tile
    const int laneH = lane >> 4;   // lane-half per ISA fragment layout
    const int m0    = wave * 16;

    v8f acc[4];
    #pragma unroll
    for (int i = 0; i < 4; ++i) acc[i] = (v8f){};

    const __bf16* aRow = sA + (m0 + laneM) * K;
    #pragma unroll
    for (int ks = 0; ks < K; ks += 32) {
        v16bf va = load_frag_a(aRow, ks, laneH);
        #pragma unroll
        for (int nt = 0; nt < 4; ++nt) {
            v16bf vb = load_frag_b(sW + (nt * 16 + laneM) * K + ks + laneH * 16);
            acc[nt] = __builtin_amdgcn_wmma_f32_16x16x32_bf16(
                false, va, false, vb, (short)0, acc[nt], false, false);
        }
    }

    // D layout: element r -> row M = laneH*8 + r, col N = laneM.
    #pragma unroll
    for (int nt = 0; nt < 4; ++nt) {
        int col = nt * 16 + laneM;
        float b = bias[col];
        #pragma unroll
        for (int r = 0; r < 8; ++r) {
            int row = R + m0 + laneH * 8 + r;
            if (row < n_rows) out[(long long)row * 64 + col] = acc[nt][r] + b;
        }
    }
}

// ---------------------------------------------------------------------------
// Column statistics: stats[c] += sum, stats[64+c] += sum of squares.
// ---------------------------------------------------------------------------
__global__ __launch_bounds__(256)
void colstats_kernel(const float* __restrict__ pre, int n_rows,
                     float* __restrict__ stats) {
    __shared__ float ss[256];
    __shared__ float ss2[256];
    int t = threadIdx.x;
    int c = t & 63;
    int rsub = t >> 6;  // 0..3
    float s = 0.0f, s2 = 0.0f;
    for (int r = blockIdx.x * 4 + rsub; r < n_rows; r += gridDim.x * 4) {
        float v = pre[(long long)r * 64 + c];
        s += v;
        s2 += v * v;
    }
    ss[t] = s;
    ss2[t] = s2;
    __syncthreads();
    if (t < 64) {
        s  = ss[t]  + ss[t + 64]  + ss[t + 128]  + ss[t + 192];
        s2 = ss2[t] + ss2[t + 64] + ss2[t + 128] + ss2[t + 192];
        atomicAdd(&stats[c], s);
        atomicAdd(&stats[64 + c], s2);
    }
}

// coef[c] = gamma*rsqrt(var+eps); coef[64+c] = beta - mean*coef[c]
__global__ void bn_finalize_kernel(const float* __restrict__ stats,
                                   const float* __restrict__ gamma,
                                   const float* __restrict__ beta,
                                   float inv_n, float* __restrict__ coef) {
    int c = threadIdx.x;
    if (c < 64) {
        float mean = stats[c] * inv_n;
        float var  = stats[64 + c] * inv_n - mean * mean;
        float a = gamma[c] * rsqrtf(var + 1e-5f);
        coef[c] = a;
        coef[64 + c] = beta[c] - mean * a;
    }
}

// In-place final BN + ReLU.
__global__ void bn_relu_inplace_kernel(float* __restrict__ data,
                                       const float* __restrict__ coef,
                                       int total) {
    int i = blockIdx.x * blockDim.x + threadIdx.x;
    if (i < total) {
        int c = i & 63;
        data[i] = fmaxf(coef[c] * data[i] + coef[64 + c], 0.0f);
    }
}

// ---------------------------------------------------------------------------
extern "C" void kernel_launch(void* const* d_in, const int* in_sizes, int n_in,
                              void* d_out, int out_size, void* d_ws, size_t ws_size,
                              hipStream_t stream) {
    const float* x     = (const float*)d_in[0];
    const float* battr = (const float*)d_in[1];
    const int*   upidx = (const int*)d_in[2];
    const int*   bidx  = (const int*)d_in[3];
    const float* eps1  = (const float*)d_in[4];
    const float* eps2  = (const float*)d_in[5];
    const float* uw1 = (const float*)d_in[6];
    const float* ub1 = (const float*)d_in[7];
    const float* ug1 = (const float*)d_in[8];
    const float* ube1 = (const float*)d_in[9];
    const float* uw2 = (const float*)d_in[10];
    const float* ub2 = (const float*)d_in[11];
    const float* ug2 = (const float*)d_in[12];
    const float* ube2 = (const float*)d_in[13];
    const float* bw1 = (const float*)d_in[14];
    const float* bb1 = (const float*)d_in[15];
    const float* bg1 = (const float*)d_in[16];
    const float* bbe1 = (const float*)d_in[17];
    const float* bw2 = (const float*)d_in[18];
    const float* bb2 = (const float*)d_in[19];
    const float* bg2 = (const float*)d_in[20];
    const float* bbe2 = (const float*)d_in[21];
    const float* cw  = (const float*)d_in[22];
    const float* cb  = (const float*)d_in[23];
    const float* cg  = (const float*)d_in[24];
    const float* cbe = (const float*)d_in[25];

    const int N  = in_sizes[0] / HD;
    const int EU = in_sizes[2] / 2;
    const int EB = in_sizes[3] / 2;
    const long long NH = (long long)N * HD;
    const float inv_n = 1.0f / (float)N;

    float* ws = (float*)d_ws;
    float* B0 = ws;                 // agg_up -> pre2_up
    float* B1 = ws + NH;            // agg_b  -> pre2_b
    float* stats = ws + 2 * NH;     // 5 * 128 floats
    float* coef  = stats + 5 * 128; // 5 * 128 floats
    float* OUT = (float*)d_out;     // pre1_up / pre1_b / pre_c / final

    const int thr = 256;
    const int gNH = (int)((NH + thr - 1) / thr);
    const int gGemm = (N + 127) / 128;

    // Phase 0: init aggregates and zero stats.
    init_agg_kernel<<<gNH, thr, 0, stream>>>(x, eps1, eps2, B0, B1, (int)NH);
    zero_kernel<<<(5 * 128 + thr - 1) / thr, thr, 0, stream>>>(stats, 5 * 128);

    // Phase 1: scatter-add aggregations.
    // agg_up[up[0][e]] += x[up[1][e]]
    {
        long long tot = (long long)EU * 16;
        scatter_kernel<<<(int)((tot + thr - 1) / thr), thr, 0, stream>>>(
            x, upidx + EU, upidx, EU, B0);
    }
    // agg_b[bidx[1][e]] += boundary_attr[bidx[0][e]]
    {
        long long tot = (long long)EB * 16;
        scatter_kernel<<<(int)((tot + thr - 1) / thr), thr, 0, stream>>>(
            battr, bidx, bidx + EB, EB, B1);
    }

    // Phase 2: up-MLP.
    gemm_bn_wmma_kernel<64><<<gGemm, thr, 0, stream>>>(
        B0, nullptr, nullptr, nullptr, uw1, ub1, OUT, N);              // pre1_up
    colstats_kernel<<<256, thr, 0, stream>>>(OUT, N, stats + 0);
    bn_finalize_kernel<<<1, 64, 0, stream>>>(stats + 0, ug1, ube1, inv_n, coef + 0);
    gemm_bn_wmma_kernel<64><<<gGemm, thr, 0, stream>>>(
        OUT, coef + 0, nullptr, nullptr, uw2, ub2, B0, N);             // pre2_up
    colstats_kernel<<<256, thr, 0, stream>>>(B0, N, stats + 128);
    bn_finalize_kernel<<<1, 64, 0, stream>>>(stats + 128, ug2, ube2, inv_n, coef + 128);

    // Phase 3: boundary-MLP.
    gemm_bn_wmma_kernel<64><<<gGemm, thr, 0, stream>>>(
        B1, nullptr, nullptr, nullptr, bw1, bb1, OUT, N);              // pre1_b
    colstats_kernel<<<256, thr, 0, stream>>>(OUT, N, stats + 256);
    bn_finalize_kernel<<<1, 64, 0, stream>>>(stats + 256, bg1, bbe1, inv_n, coef + 256);
    gemm_bn_wmma_kernel<64><<<gGemm, thr, 0, stream>>>(
        OUT, coef + 256, nullptr, nullptr, bw2, bb2, B1, N);           // pre2_b
    colstats_kernel<<<256, thr, 0, stream>>>(B1, N, stats + 384);
    bn_finalize_kernel<<<1, 64, 0, stream>>>(stats + 384, bg2, bbe2, inv_n, coef + 384);

    // Phase 4: combine Linear(2H->H) with fused BN+ReLU of both halves.
    gemm_bn_wmma_kernel<128><<<gGemm, thr, 0, stream>>>(
        B0, coef + 128, B1, coef + 384, cw, cb, OUT, N);               // pre_c
    colstats_kernel<<<256, thr, 0, stream>>>(OUT, N, stats + 512);
    bn_finalize_kernel<<<1, 64, 0, stream>>>(stats + 512, cg, cbe, inv_n, coef + 512);
    bn_relu_inplace_kernel<<<gNH, thr, 0, stream>>>(OUT, coef + 512, (int)NH);
}
